// Decoder_59820304499127
// MI455X (gfx1250) — matrix operand order repaired and checked by reading, not compiled, not run on hardware
//
#include <hip/hip_runtime.h>
#include <stdint.h>

// ---------------------------------------------------------------------------
// ESIM-style NLI decoder for MI455X (gfx1250).
// All GEMMs: v_wmma_f32_16x16x32_bf16 (wave32), f32 accumulation.
// MLP GEMM stages A via async global->LDS (ASYNCcnt) and B via the Tensor
// Data Mover (TENSORcnt) with TDM LDS padding for bank-friendly ds_load_b128.
// Shapes: I=J=256, B=64, H=1024, NHID=1024, NCLS=3.
// ---------------------------------------------------------------------------

typedef __attribute__((ext_vector_type(16))) __bf16 v16bf;
typedef __attribute__((ext_vector_type(8)))  __bf16 bf16x8;
typedef __attribute__((ext_vector_type(4)))  __bf16 bf16x4;
typedef __attribute__((ext_vector_type(8)))  float  v8f;
typedef __attribute__((ext_vector_type(4)))  float  f32x4;
typedef __attribute__((ext_vector_type(4)))  unsigned int u32x4;
typedef __attribute__((ext_vector_type(8)))  int    i32x8;
typedef __attribute__((ext_vector_type(4)))  int    i32x4;

// Types matching the async-LDS builtin prototype: pointer to GCC-style
// <4 x i32> in global (AS1) / LDS (AS3) address spaces.
typedef int v4i_vs __attribute__((vector_size(16)));
typedef __attribute__((address_space(1))) v4i_vs* gptr_v4i;
typedef __attribute__((address_space(3))) v4i_vs* lptr_v4i;

#define POS 256      // I == J
#define BAT 64
#define HD  1024

#if __has_builtin(__builtin_amdgcn_global_load_async_to_lds_b128)
#define HAVE_ASYNC_LDS 1
#else
#define HAVE_ASYNC_LDS 0
#endif
#if __has_builtin(__builtin_amdgcn_tensor_load_to_lds)
#define HAVE_TDM 1
#else
#define HAVE_TDM 0
#endif

__device__ __forceinline__ v16bf cat8(bf16x8 a, bf16x8 b) {
  return __builtin_shufflevector(a, b, 0,1,2,3,4,5,6,7,8,9,10,11,12,13,14,15);
}
__device__ __forceinline__ v8f vzero8() {
  v8f z = {0.f,0.f,0.f,0.f,0.f,0.f,0.f,0.f};
  return z;
}
__device__ __forceinline__ v8f wmma_bf16(v16bf a, v16bf b, v8f c) {
  return __builtin_amdgcn_wmma_f32_16x16x32_bf16(false, a, false, b,
                                                 (short)0, c, false, false);
}

#if HAVE_ASYNC_LDS
__device__ __forceinline__ void async_g2l_b128(const void* g, void* l) {
  __builtin_amdgcn_global_load_async_to_lds_b128(
      (gptr_v4i)(uintptr_t)g,
      (lptr_v4i)(uint32_t)(uintptr_t)l, 0, 0);
}
__device__ __forceinline__ void wait_async0() {
#if __has_builtin(__builtin_amdgcn_s_wait_asynccnt)
  __builtin_amdgcn_s_wait_asynccnt(0);
#else
  asm volatile("s_wait_asynccnt 0x0" ::: "memory");
#endif
}
#endif

#if HAVE_TDM
// 2D tile load: tile_n rows x tile_k halves (2B elems), row stride strideK
// halves, into LDS at lds_off. TDM pad: +4 DWORDs every 16 DWORDs so a
// 32-half row lands on an 80B (40-half) LDS stride.
__device__ __forceinline__ void tdm_load_2d(uint32_t lds_off, uint64_t gaddr,
                                            int tile_k, int tile_n, int strideK) {
  u32x4 g0;
  g0[0] = 1u;                                       // count=1
  g0[1] = lds_off;                                  // lds_addr (bytes)
  g0[2] = (uint32_t)(gaddr & 0xFFFFFFFFu);          // global_addr[31:0]
  g0[3] = (uint32_t)((gaddr >> 32) & 0x1FFFFFFu) | (2u << 30); // [56:32]|type=2
  i32x8 g1;
  g1[0] = (1 << 16) | (1 << 20) | (3 << 22) | (3 << 25); // 2B elems, pad 4dw/16dw
  g1[1] = tile_k << 16;                             // tensor_dim0[15:0]
  g1[2] = tile_n << 16;                             // tensor_dim1[15:0]
  g1[3] = tile_k << 16;                             // tile_dim0
  g1[4] = tile_n;                                   // tile_dim1 (tile_dim2=0)
  g1[5] = strideK;                                  // tensor_dim0_stride[31:0]
  g1[6] = 0;
  g1[7] = 0;
  i32x4 z4 = {0, 0, 0, 0};
#if __clang_major__ >= 23
  i32x8 z8 = {0, 0, 0, 0, 0, 0, 0, 0};
  __builtin_amdgcn_tensor_load_to_lds(g0, g1, z4, z4, z8, 0);
#else
  __builtin_amdgcn_tensor_load_to_lds(g0, g1, z4, z4, 0);
#endif
}
__device__ __forceinline__ void wait_tensor0() {
#if __has_builtin(__builtin_amdgcn_s_wait_tensorcnt)
  __builtin_amdgcn_s_wait_tensorcnt(0);
#else
  asm volatile("s_wait_tensorcnt 0x0" ::: "memory");
#endif
}
#endif

// ---------------------------------------------------------------------------
// Elementwise f32 -> bf16 (flat)
// ---------------------------------------------------------------------------
__global__ void cvt_bf16_kernel(const float* __restrict__ src,
                                __bf16* __restrict__ dst, int n4) {
  int i = blockIdx.x * blockDim.x + threadIdx.x;
  if (i >= n4) return;
  f32x4 v = *(const f32x4*)(src + (size_t)i * 4);
  bf16x4 o;
  o[0] = (__bf16)v[0]; o[1] = (__bf16)v[1];
  o[2] = (__bf16)v[2]; o[3] = (__bf16)v[3];
  *(bf16x4*)(dst + (size_t)i * 4) = o;
}

// f32 emb [pos,b,h] -> bf16 into X_all[(rowbase + pos*B + b)][0:1024]
__global__ void cvt_emb_kernel(const float* __restrict__ src,
                               __bf16* __restrict__ X, int rowbase, int n4) {
  int i = blockIdx.x * blockDim.x + threadIdx.x;
  if (i >= n4) return;
  size_t e   = (size_t)i * 4;
  size_t row = e >> 10;
  int    col = (int)(e & 1023);
  f32x4 v = *(const f32x4*)(src + e);
  bf16x4 o;
  o[0] = (__bf16)v[0]; o[1] = (__bf16)v[1];
  o[2] = (__bf16)v[2]; o[3] = (__bf16)v[3];
  *(bf16x4*)(X + ((size_t)rowbase + row) * 2048 + col) = o;
}

// W f32 [K][1024] -> bf16 Wt [1024][K]
__global__ void transpose_w_kernel(const float* __restrict__ W,
                                   __bf16* __restrict__ Wt, int K) {
  int i = blockIdx.x * blockDim.x + threadIdx.x;
  if (i >= K * 1024) return;
  int k = i >> 10;
  int n = i & 1023;
  Wt[(size_t)n * K + k] = (__bf16)W[i];
}

// ---------------------------------------------------------------------------
// Scores: S[b,i,j] = sum_h Penc16[i,b,h] * Henc16[j,b,h]
// One wave per 16x16 tile; both operands K-contiguous from global.
// ---------------------------------------------------------------------------
__global__ void scores_kernel(const __bf16* __restrict__ Penc,
                              const __bf16* __restrict__ Henc,
                              float* __restrict__ Sjbi,   // [b][j][i]
                              float* __restrict__ Sibj) { // [b][i][j]
  const int lane = threadIdx.x;
  const int li = lane & 15;
  const int hi = lane >> 4;
  const int it = blockIdx.x * 16;
  const int jt = blockIdx.y * 16;
  const int b  = blockIdx.z;

  const __bf16* arow = Penc + ((size_t)(it + li) * BAT + b) * HD;
  const __bf16* brow = Henc + ((size_t)(jt + li) * BAT + b) * HD;

  v8f acc = vzero8();
  for (int kb = 0; kb < HD; kb += 32) {
    const __bf16* ap = arow + kb + (hi ? 8 : 0);
    v16bf A = cat8(*(const bf16x8*)ap, *(const bf16x8*)(ap + 16));
    const __bf16* bp = brow + kb + (hi ? 8 : 0);
    v16bf Bm = cat8(*(const bf16x8*)bp, *(const bf16x8*)(bp + 16));
    acc = wmma_bf16(A, Bm, acc);
  }

  const int jc = jt + li;
  const int mb = it + (hi ? 8 : 0);
  float* pj = Sjbi + ((size_t)b * POS + jc) * POS + mb;
  f32x4 lo  = __builtin_shufflevector(acc, acc, 0,1,2,3);
  f32x4 hiv = __builtin_shufflevector(acc, acc, 4,5,6,7);
  *(f32x4*)pj = lo;
  *(f32x4*)(pj + 4) = hiv;
  float* pi = Sibj + ((size_t)b * POS + mb) * POS + jc;
#pragma unroll
  for (int v = 0; v < 8; ++v) pi[(size_t)v * POS] = acc[v];
}

// ---------------------------------------------------------------------------
// Masked softmax, one wave per 256-f32 row; mask[q*BAT + b] (bool byte).
// ---------------------------------------------------------------------------
__global__ void softmax_kernel(const float* __restrict__ S,
                               const unsigned char* __restrict__ mask,
                               __bf16* __restrict__ out) {
  const int lane = threadIdx.x & 31;
  const int wid  = (blockIdx.x * blockDim.x + threadIdx.x) >> 5;
  const int b    = wid >> 8;
  const float* row = S + (size_t)wid * POS;
  const int q0 = lane * 8;

  f32x4 v0 = *(const f32x4*)(row + q0);
  f32x4 v1 = *(const f32x4*)(row + q0 + 4);
  float x[8];
#pragma unroll
  for (int e = 0; e < 8; ++e) {
    float v = (e < 4) ? v0[e] : v1[e - 4];
    x[e] = mask[(q0 + e) * BAT + b] ? -__builtin_inff() : v;
  }
  float mx = x[0];
#pragma unroll
  for (int e = 1; e < 8; ++e) mx = fmaxf(mx, x[e]);
  for (int off = 16; off >= 1; off >>= 1) mx = fmaxf(mx, __shfl_xor(mx, off, 32));
  float sum = 0.f, ex[8];
#pragma unroll
  for (int e = 0; e < 8; ++e) { ex[e] = __expf(x[e] - mx); sum += ex[e]; }
  for (int off = 16; off >= 1; off >>= 1) sum += __shfl_xor(sum, off, 32);
  float inv = 1.0f / sum;
  bf16x8 o;
#pragma unroll
  for (int e = 0; e < 8; ++e) o[e] = (__bf16)(ex[e] * inv);
  *(bf16x8*)(out + (size_t)wid * POS + q0) = o;
}

// ---------------------------------------------------------------------------
// Context GEMM: C[pos,h] = sum_k attnT[b,pos,k] * Emb[k,b,h]; manual LDS
// transpose staging of B; epilogue mask->0, write bf16 into X[:,1024:2048].
// ---------------------------------------------------------------------------
#define LDSB_STRIDE 40  // 32 K-halves + 8 pad (80B rows, 16B aligned)

__global__ void ctx_gemm_kernel(const __bf16* __restrict__ attnT,
                                const __bf16* __restrict__ X,
                                int rowbaseB, int rowbaseOut,
                                const unsigned char* __restrict__ mask,
                                __bf16* __restrict__ Xout) {
  __shared__ __align__(16) __bf16 ldsB[128 * LDSB_STRIDE];
  const int tid  = threadIdx.x;
  const int lane = tid & 31;
  const int w    = tid >> 5;
  const int li   = lane & 15;
  const int hi   = lane >> 4;
  const int m0   = blockIdx.x * 64;
  const int n0   = blockIdx.y * 128;
  const int b    = blockIdx.z;

  v8f acc[4] = {vzero8(), vzero8(), vzero8(), vzero8()};
  const __bf16* Abase = attnT + (size_t)b * POS * POS;

  for (int kb = 0; kb < POS; kb += 32) {
    {
      const int k  = tid >> 3;
      const int nn = (tid & 7) * 16;
      const __bf16* src =
          X + ((size_t)rowbaseB + (size_t)(kb + k) * BAT + b) * 2048 + (n0 + nn);
      bf16x8 s0 = *(const bf16x8*)src;
      bf16x8 s1 = *(const bf16x8*)(src + 8);
#pragma unroll
      for (int e = 0; e < 8; ++e) {
        ldsB[(nn + e) * LDSB_STRIDE + k]     = s0[e];
        ldsB[(nn + 8 + e) * LDSB_STRIDE + k] = s1[e];
      }
    }
    __syncthreads();
    const __bf16* bp = &ldsB[(w * 16 + li) * LDSB_STRIDE + (hi ? 8 : 0)];
    v16bf Bop = cat8(*(const bf16x8*)bp, *(const bf16x8*)(bp + 16));
#pragma unroll
    for (int s = 0; s < 4; ++s) {
      const __bf16* ap = Abase + (size_t)(m0 + s * 16 + li) * POS + kb + (hi ? 8 : 0);
      v16bf A = cat8(*(const bf16x8*)ap, *(const bf16x8*)(ap + 16));
      acc[s] = wmma_bf16(A, Bop, acc[s]);
    }
    __syncthreads();
  }

  const int n = n0 + w * 16 + li;
#pragma unroll
  for (int s = 0; s < 4; ++s) {
#pragma unroll
    for (int v = 0; v < 8; ++v) {
      int pos = m0 + s * 16 + v + (hi ? 8 : 0);
      float val = mask[pos * BAT + b] ? 0.f : acc[s][v];
      Xout[((size_t)rowbaseOut + (size_t)pos * BAT + b) * 2048 + 1024 + n] =
          (__bf16)val;
    }
  }
}

// ---------------------------------------------------------------------------
// MLP GEMM: out16[M][1024] = relu( A[M][K] x Bt[1024][K]^T + bias ), bf16 in,
// f32 accum. A tile staged via async global->LDS (ASYNCcnt); B tile staged
// via the Tensor Data Mover (TENSORcnt) with padded LDS rows.
// block = 256 threads (8 waves); tile 64 x 128; grid (M/64, 8).
// ---------------------------------------------------------------------------
__global__ void mlp_gemm_kernel(const __bf16* __restrict__ A,
                                const __bf16* __restrict__ Bt,
                                const float* __restrict__ bias,
                                int K,
                                __bf16* __restrict__ out16) {
  __shared__ __align__(16) __bf16 ldsA[64 * LDSB_STRIDE];
  __shared__ __align__(16) __bf16 ldsB[128 * LDSB_STRIDE];
  const int tid  = threadIdx.x;
  const int lane = tid & 31;
  const int w    = tid >> 5;
  const int li   = lane & 15;
  const int hi   = lane >> 4;
  const int m0   = blockIdx.x * 64;
  const int n0   = blockIdx.y * 128;

  v8f acc[4] = {vzero8(), vzero8(), vzero8(), vzero8()};

  for (int kb = 0; kb < K; kb += 32) {
    // ---- stage A tile (64 x 32 halves) ----
    {
      const int row = tid >> 2;
      const int kh  = (tid & 3) * 8;
      const __bf16* g = A + (size_t)(m0 + row) * K + kb + kh;
      __bf16* l = &ldsA[row * LDSB_STRIDE + kh];
#if HAVE_ASYNC_LDS
      async_g2l_b128(g, l);
#else
      *(bf16x8*)l = *(const bf16x8*)g;
#endif
    }
    // ---- stage B tile (128 rows x 32 halves, row stride K) ----
#if HAVE_TDM
    if (w == 0) {
      tdm_load_2d((uint32_t)(uintptr_t)&ldsB[0],
                  (uint64_t)(uintptr_t)(Bt + (size_t)n0 * K + kb),
                  /*tile_k=*/32, /*tile_n=*/128, /*strideK=*/K);
      wait_tensor0();
    }
#else
    {
      const int row = tid >> 1;
      const int kh  = (tid & 1) * 16;
      const __bf16* g = Bt + (size_t)(n0 + row) * K + kb + kh;
      __bf16* l = &ldsB[row * LDSB_STRIDE + kh];
      *(bf16x8*)l       = *(const bf16x8*)g;
      *(bf16x8*)(l + 8) = *(const bf16x8*)(g + 8);
    }
#endif
#if HAVE_ASYNC_LDS
    wait_async0();
#endif
    __syncthreads();

    // ---- compute from LDS ----
    const __bf16* bp = &ldsB[(w * 16 + li) * LDSB_STRIDE + (hi ? 8 : 0)];
    v16bf Bop = cat8(*(const bf16x8*)bp, *(const bf16x8*)(bp + 16));
#pragma unroll
    for (int s = 0; s < 4; ++s) {
      const __bf16* ap = &ldsA[(s * 16 + li) * LDSB_STRIDE + (hi ? 8 : 0)];
      v16bf Aop = cat8(*(const bf16x8*)ap, *(const bf16x8*)(ap + 16));
      acc[s] = wmma_bf16(Aop, Bop, acc[s]);
    }
    __syncthreads();
  }

  const int n = n0 + w * 16 + li;
  const float bv = bias[n];
#pragma unroll
  for (int s = 0; s < 4; ++s) {
#pragma unroll
    for (int v = 0; v < 8; ++v) {
      int row = m0 + s * 16 + v + (hi ? 8 : 0);
      float val = fmaxf(acc[s][v] + bv, 0.f);
      out16[(size_t)row * 1024 + n] = (__bf16)val;
    }
  }
}

// ---------------------------------------------------------------------------
// Deterministic pooling: pool16[b][side*1024 + n] = sum_pos comp16[row][n]
// ---------------------------------------------------------------------------
__global__ void pool_reduce_kernel(const __bf16* __restrict__ comp16,
                                   __bf16* __restrict__ pool16) {
  int t = blockIdx.x * blockDim.x + threadIdx.x;
  if (t >= BAT * 2048) return;
  int n    = t & 1023;
  int side = (t >> 10) & 1;
  int b    = t >> 11;
  float s = 0.f;
  const __bf16* base = comp16 + ((size_t)side * 16384 + b) * 1024 + n;
  for (int pos = 0; pos < POS; ++pos)
    s += (float)base[(size_t)pos * BAT * 1024];
  pool16[(size_t)b * 2048 + side * 1024 + n] = (__bf16)s;
}

__global__ void final_kernel(const __bf16* __restrict__ aggh,
                             const float* __restrict__ aW2,
                             const float* __restrict__ ab2,
                             float* __restrict__ out) {
  int t = threadIdx.x;
  if (t >= BAT * 3) return;
  int b = t / 3, c = t % 3;
  float s = ab2[c];
  const __bf16* row = aggh + (size_t)b * 1024;
  for (int k = 0; k < 1024; ++k) s += (float)row[k] * aW2[k * 3 + c];
  out[b * 3 + c] = s;
}

// ---------------------------------------------------------------------------
// Host side
// ---------------------------------------------------------------------------
extern "C" void kernel_launch(void* const* d_in, const int* in_sizes, int n_in,
                              void* d_out, int out_size, void* d_ws, size_t ws_size,
                              hipStream_t stream) {
  (void)in_sizes; (void)n_in; (void)out_size; (void)ws_size;

  const float* P_enc = (const float*)d_in[0];
  const float* H_enc = (const float*)d_in[1];
  const float* P_emb = (const float*)d_in[2];
  const float* H_emb = (const float*)d_in[3];
  const unsigned char* prem_mask = (const unsigned char*)d_in[4];
  const unsigned char* hypo_mask = (const unsigned char*)d_in[5];
  const float* cW1 = (const float*)d_in[6];
  const float* cb1 = (const float*)d_in[7];
  const float* cW2 = (const float*)d_in[8];
  const float* cb2 = (const float*)d_in[9];
  const float* aW1 = (const float*)d_in[10];
  const float* ab1 = (const float*)d_in[11];
  const float* aW2 = (const float*)d_in[12];
  const float* ab2 = (const float*)d_in[13];
  float* out = (float*)d_out;

  char* w = (char*)d_ws;
  const size_t N_ENC = (size_t)POS * BAT * HD;
  __bf16* Penc16 = (__bf16*)(w + 0);
  __bf16* Henc16 = (__bf16*)(w + 33554432);
  float*  Sjbi   = (float*) (w + 67108864);
  float*  Sibj   = (float*) (w + 83886080);
  __bf16* PattnT = (__bf16*)(w + 100663296);
  __bf16* HattnT = (__bf16*)(w + 109051904);
  __bf16* X_all  = (__bf16*)(w + 117440512);   // [32768][2048]
  __bf16* comp16 = X_all;                      // alias: X dead after L1
  __bf16* h1     = (__bf16*)(w + 251658240);   // [32768][1024]
  __bf16* W1T    = (__bf16*)(w + 318767104);   // [1024][2048]
  __bf16* W2T    = (__bf16*)(w + 322961408);   // [1024][1024]
  __bf16* aW1T   = (__bf16*)(w + 325058560);   // [1024][2048]
  __bf16* pool16 = (__bf16*)(w + 329252864);   // [64][2048]
  __bf16* aggh16 = (__bf16*)(w + 329515008);   // [64][1024]

  const int n4 = (int)(N_ENC / 4);
  cvt_bf16_kernel<<<(n4 + 255) / 256, 256, 0, stream>>>(P_enc, Penc16, n4);
  cvt_bf16_kernel<<<(n4 + 255) / 256, 256, 0, stream>>>(H_enc, Henc16, n4);
  cvt_emb_kernel<<<(n4 + 255) / 256, 256, 0, stream>>>(H_emb, X_all, 0, n4);
  cvt_emb_kernel<<<(n4 + 255) / 256, 256, 0, stream>>>(P_emb, X_all, 16384, n4);
  transpose_w_kernel<<<(2048 * 1024) / 256, 256, 0, stream>>>(cW1, W1T, 2048);
  transpose_w_kernel<<<(1024 * 1024) / 256, 256, 0, stream>>>(cW2, W2T, 1024);
  transpose_w_kernel<<<(2048 * 1024) / 256, 256, 0, stream>>>(aW1, aW1T, 2048);

  scores_kernel<<<dim3(16, 16, 64), 32, 0, stream>>>(Penc16, Henc16, Sjbi, Sibj);

  softmax_kernel<<<2048, 256, 0, stream>>>(Sjbi, prem_mask, PattnT);
  softmax_kernel<<<2048, 256, 0, stream>>>(Sibj, hypo_mask, HattnT);

  ctx_gemm_kernel<<<dim3(4, 8, 64), 256, 0, stream>>>(
      PattnT, X_all, 16384, 0, hypo_mask, X_all);
  ctx_gemm_kernel<<<dim3(4, 8, 64), 256, 0, stream>>>(
      HattnT, X_all, 0, 16384, prem_mask, X_all);

  mlp_gemm_kernel<<<dim3(512, 8), 256, 0, stream>>>(X_all, W1T, cb1, 2048, h1);
  mlp_gemm_kernel<<<dim3(512, 8), 256, 0, stream>>>(h1, W2T, cb2, 1024, comp16);

  pool_reduce_kernel<<<(BAT * 2048) / 256, 256, 0, stream>>>(comp16, pool16);
  mlp_gemm_kernel<<<dim3(1, 8), 256, 0, stream>>>(pool16, aW1T, ab1, 2048, aggh16);
  final_kernel<<<1, 256, 0, stream>>>(aggh16, aW2, ab2, out);
}
